// DummyModel_78383153152447
// MI455X (gfx1250) — compile-verified, matrix operand order checked by reading, not run on hardware
//
#include <hip/hip_runtime.h>
#include <hip/hip_bf16.h>
#include <math.h>

// Problem constants (match reference)
#define BB 16
#define TT 256
#define UU 64
#define U1 65
#define HH 512
#define G4 2048   // 4*H
#define VJ 128    // DEC_VOCAB+1 (joint vocab)

typedef __attribute__((ext_vector_type(16))) _Float16 v16h;
typedef __attribute__((ext_vector_type(8)))  float    v8f;

union AFrag { v16h v; unsigned u[8]; _Float16 h[16]; };
union CFrag { v8f v; float f[8]; };

// ISA 7.12.2: 16-bit A/B 16x32 fragment: lane l holds row (l&15); VGPR v holds
// K pair starting at 2*((v&3) + (v>>2)*8 + (l>>4)*4).
__device__ __forceinline__ int kpair(int v, int lh) {
  return 2 * ((v & 3) + ((v >> 2) << 3) + (lh << 2));
}

__device__ __forceinline__ float sigf(float x) { return 1.f / (1.f + __expf(-x)); }

__device__ __forceinline__ void grid_barrier(unsigned* ctrl, unsigned* localSense, unsigned nblk) {
  __syncthreads();
  if (threadIdx.x == 0) {
    unsigned s = *localSense ^ 1u;
    *localSense = s;
    __threadfence();
    if (atomicAdd(&ctrl[0], 1u) == nblk - 1u) {
      ctrl[0] = 0u;
      __threadfence();
      atomicExch(&ctrl[1], s);
    } else {
      while (atomicAdd(&ctrl[1], 0u) != s) { }
    }
  }
  __syncthreads();
}

// ---------------------------------------------------------------------------
// Prep: transpose+convert W (K x N, f32) -> Wt (N x K, f16)
__global__ void f16t_kernel(const float* __restrict__ W, _Float16* __restrict__ Wt,
                            int K, int N) {
  int idx = blockIdx.x * blockDim.x + threadIdx.x;
  if (idx < N * K) {
    int n = idx / K, k = idx % K;
    Wt[idx] = (_Float16)W[(size_t)k * N + n];
  }
}

// zx[b,t,:] = Wk[token(b,t), :] + bias   (one-hot matmul == row gather)
__global__ void zx_kernel(const int* __restrict__ seq, const float* __restrict__ Wk,
                          const float* __restrict__ bias, float* __restrict__ zx,
                          int steps, int isDec) {
  int bt = blockIdx.x;
  int b = bt / steps, t = bt % steps;
  int tok;
  if (isDec) tok = (t == 0) ? 0 : (seq[b * UU + (t - 1)] + 1);
  else       tok = seq[b * steps + t];
  const float* row = Wk + (size_t)tok * G4;
  float* dst = zx + (size_t)bt * G4;
  for (int c = threadIdx.x; c < G4; c += blockDim.x) dst[c] = row[c] + bias[c];
}

// ---------------------------------------------------------------------------
// Persistent LSTM: 16 WGs x 256 thr. WG g owns hidden slice [32g, 32g+32) i.e.
// z-columns {gate*512 + 32g + [0,32)} for 4 gates = 8 N-tiles, one per wave.
// Each wave keeps its 16x512 f16 slice of Wr^T in 16 v16h register fragments.
__global__ __launch_bounds__(256, 1) void lstm_kernel(
    const float* __restrict__ zx,      // [B, steps, 2048]
    const _Float16* __restrict__ Wrt,  // [2048, 512] (row = output column)
    const int* __restrict__ lens, int lenAdd, int steps,
    float* __restrict__ hstate, float* __restrict__ cstate,   // [B,512]
    _Float16* __restrict__ hbuf,       // [2][B*512] double-buffered f16 h
    _Float16* __restrict__ hs_out,     // [B, steps, 512]
    unsigned* ctrl) {
  const int g = blockIdx.x;
  const int tid = threadIdx.x;
  const int w = tid >> 5;
  const int l = tid & 31;
  const int lm = l & 15;
  const int lh = l >> 4;

  __shared__ __align__(16) _Float16 ldsH[BB * HH];   // 16 KB
  __shared__ float ldsZ[BB * 128];                   // 8 KB
  __shared__ unsigned localSense;
  if (tid == 0) localSense = 0;

  // Register-resident B fragments: wave's 16 output columns, full K=512.
  const int col0 = (w >> 1) * HH + 32 * g + 16 * (w & 1);
  AFrag bfr[16];
#pragma unroll
  for (int kt = 0; kt < 16; ++kt) {
    const _Float16* wrow = Wrt + (size_t)(col0 + lm) * HH + kt * 32;
#pragma unroll
    for (int v = 0; v < 8; ++v)
      bfr[kt].u[v] = *(const unsigned*)(wrow + kpair(v, lh));
  }

  // Zero this WG's state slice + h f16 buffer 0.
  for (int k = 0; k < 2; ++k) {
    int idx = tid + k * 256;
    int b = idx >> 5, jl = idx & 31, j = 32 * g + jl;
    hstate[b * HH + j] = 0.f;
    cstate[b * HH + j] = 0.f;
    hbuf[b * HH + j] = (_Float16)0.f;
  }
  __threadfence();
  grid_barrier(ctrl, &localSense, BB);

  for (int t = 0; t < steps; ++t) {
    const int rb = t & 1;
    const _Float16* hin = hbuf + (size_t)rb * (BB * HH);
    // broadcast h into LDS (16 KB, 16B vector copies)
    for (int i = tid; i < (BB * HH) / 8; i += 256)
      ((float4*)ldsH)[i] = ((const float4*)hin)[i];
    if (t + 1 < steps)
      __builtin_prefetch(&zx[(size_t)(lm * steps + t + 1) * G4 + col0], 0, 1);
    __syncthreads();

    // C initialized with zx tile (f32 16x16: VGPR r -> row r + 8*(l>>4), col l&15)
    CFrag c;
#pragma unroll
    for (int r = 0; r < 8; ++r) {
      int mb = r + (lh << 3);
      c.f[r] = zx[(size_t)(mb * steps + t) * G4 + col0 + lm];
    }
#pragma unroll
    for (int kt = 0; kt < 16; ++kt) {
      AFrag a;
#pragma unroll
      for (int v = 0; v < 8; ++v)
        a.u[v] = *(const unsigned*)&ldsH[lm * HH + kt * 32 + kpair(v, lh)];
      c.v = __builtin_amdgcn_wmma_f32_16x16x32_f16(false, a.v, false, bfr[kt].v,
                                                   (short)0, c.v, false, false);
    }
#pragma unroll
    for (int r = 0; r < 8; ++r) {
      int mb = r + (lh << 3);
      ldsZ[mb * 128 + w * 16 + lm] = c.f[r];   // col = gate*32 + local-hidden
    }
    __syncthreads();

    // Gate math for this WG's 32 hidden units x 16 batches.
    _Float16* hout = hbuf + (size_t)(rb ^ 1) * (BB * HH);
    for (int k = 0; k < 2; ++k) {
      int idx = tid + k * 256;
      int b = idx >> 5, jl = idx & 31, j = 32 * g + jl;
      float zi = ldsZ[b * 128 + jl];
      float zf = ldsZ[b * 128 + 32 + jl];
      float zg = ldsZ[b * 128 + 64 + jl];
      float zo = ldsZ[b * 128 + 96 + jl];
      float cp = cstate[b * HH + j];
      float hp = hstate[b * HH + j];
      float cn = sigf(zf) * cp + sigf(zi) * tanhf(zg);
      float hn = sigf(zo) * tanhf(cn);
      bool m = t < (lens[b] + lenAdd);
      float c2 = m ? cn : cp;
      float h2 = m ? hn : hp;
      cstate[b * HH + j] = c2;
      hstate[b * HH + j] = h2;
      hout[b * HH + j] = (_Float16)h2;
      hs_out[(size_t)(b * steps + t) * HH + j] = (_Float16)h2;
    }
    __syncthreads();
    __threadfence();
    grid_barrier(ctrl, &localSense, BB);
  }
}

// ---------------------------------------------------------------------------
// Y[M,128] = X[M,512](f16) @ Wt^T (Wt is [128,512] f16, row = out col) (+bias)
__global__ __launch_bounds__(256) void proj_kernel(
    const _Float16* __restrict__ X, const _Float16* __restrict__ Wt,
    const float* __restrict__ bias, float* __restrict__ Y, int M) {
  int w = threadIdx.x >> 5, l = threadIdx.x & 31, lm = l & 15, lh = l >> 4;
  int mt = blockIdx.x, nt = w;   // 8 N-tiles of 16 = 128 cols
  CFrag c;
#pragma unroll
  for (int r = 0; r < 8; ++r) c.f[r] = 0.f;
  const _Float16* xrow = X + (size_t)(mt * 16 + lm) * HH;
  const _Float16* wrow = Wt + (size_t)(nt * 16 + lm) * HH;
#pragma unroll
  for (int kt = 0; kt < 16; ++kt) {
    AFrag a, b;
#pragma unroll
    for (int v = 0; v < 8; ++v) {
      int kb = kt * 32 + kpair(v, lh);
      a.u[v] = *(const unsigned*)(xrow + kb);
      b.u[v] = *(const unsigned*)(wrow + kb);
    }
    c.v = __builtin_amdgcn_wmma_f32_16x16x32_f16(false, a.v, false, b.v,
                                                 (short)0, c.v, false, false);
  }
#pragma unroll
  for (int r = 0; r < 8; ++r) {
    int mb = mt * 16 + r + (lh << 3);
    int nn = nt * 16 + lm;
    float val = c.f[r];
    if (bias) val += bias[nn];
    if (mb < M) Y[(size_t)mb * VJ + nn] = val;
  }
}

// ---------------------------------------------------------------------------
// Fused joint: for each (b,t,u) compute lse over v of relu(E+D) and emit only
// the blank and label log-probs. Never materializes the 34M-elem logits.
__global__ __launch_bounds__(128) void joint_kernel(
    const float* __restrict__ E,        // [B*T,128]
    const float* __restrict__ D,        // [B*U1,128] (includes joint_b)
    const int* __restrict__ dec_seqs,   // [B,64]
    float* __restrict__ blank,          // [B*T,65]
    float* __restrict__ emit) {         // [B*T,64]
  int bt = blockIdx.x;
  int b = bt / TT;
  int tid = threadIdx.x, w = tid >> 5, l = tid & 31;
  __shared__ float Eb[VJ];
  if (tid < VJ) Eb[tid] = E[(size_t)bt * VJ + tid];
  __syncthreads();
  for (int u = w; u < U1; u += 4) {
    const float* Drow = D + (size_t)(b * U1 + u) * VJ;
    float vals[4], m = -INFINITY;
#pragma unroll
    for (int i = 0; i < 4; ++i) {
      int v = l + 32 * i;
      float x = Eb[v] + Drow[v];
      x = x > 0.f ? x : 0.f;
      vals[i] = x;
      m = fmaxf(m, x);
    }
    for (int off = 16; off; off >>= 1) m = fmaxf(m, __shfl_xor(m, off, 32));
    float s = 0.f;
#pragma unroll
    for (int i = 0; i < 4; ++i) s += __expf(vals[i] - m);
    for (int off = 16; off; off >>= 1) s += __shfl_xor(s, off, 32);
    float lse = m + __logf(s);
    if (l == 0) {
      float bl = Eb[0] + Drow[0]; bl = bl > 0.f ? bl : 0.f;
      blank[(size_t)bt * U1 + u] = bl - lse;
      if (u < UU) {
        int lab = dec_seqs[b * UU + u] + 1;
        float em = Eb[lab] + Drow[lab]; em = em > 0.f ? em : 0.f;
        emit[(size_t)bt * UU + u] = em - lse;
      }
    }
  }
}

// ---------------------------------------------------------------------------
// RNNT alpha recursion: 1 block per batch; serial over T, parallel scans over U.
__global__ __launch_bounds__(128) void alpha_kernel(
    const float* __restrict__ blank, const float* __restrict__ emit,
    const int* __restrict__ enc_lens, const int* __restrict__ dec_lens,
    float* __restrict__ loss) {
  int b = blockIdx.x, tid = threadIdx.x;
  __shared__ float alpha[U1];
  __shared__ float s1[U1];
  int tcap = enc_lens[b] - 1;
  int ucap = dec_lens[b];
  const float* blk = blank + (size_t)b * TT * U1;
  const float* emt = emit + (size_t)b * TT * UU;

  // alpha0 = [0, cumsum(emit[:,0,:])]
  if (tid < UU) s1[tid] = emt[tid];
  __syncthreads();
  for (int off = 1; off < UU; off <<= 1) {
    float v = 0.f; bool act = (tid < UU) && (tid >= off);
    if (act) v = s1[tid - off] + s1[tid];
    __syncthreads();
    if (act) s1[tid] = v;
    __syncthreads();
  }
  if (tid == 0) alpha[0] = 0.f;
  if (tid < UU) alpha[tid + 1] = s1[tid];
  __syncthreads();
  if (tcap == 0 && tid == 0) loss[b] = -(alpha[ucap] + blk[ucap]);

  for (int t = 1; t < TT; ++t) {
    float aval = 0.f;
    if (tid < U1) aval = alpha[tid] + blk[(size_t)(t - 1) * U1 + tid];
    if (tid < UU) s1[tid] = emt[(size_t)t * UU + tid];
    __syncthreads();
    for (int off = 1; off < UU; off <<= 1) {            // cumsum(emit_t)
      float v = 0.f; bool act = (tid < UU) && (tid >= off);
      if (act) v = s1[tid - off] + s1[tid];
      __syncthreads();
      if (act) s1[tid] = v;
      __syncthreads();
    }
    float cval = 0.f;
    if (tid < U1 && tid > 0) cval = s1[tid - 1];
    __syncthreads();
    if (tid < U1) s1[tid] = aval - cval;
    __syncthreads();
    for (int off = 1; off < U1; off <<= 1) {            // logaddexp scan
      float v = 0.f; bool act = (tid < U1) && (tid >= off);
      if (act) {
        float x = s1[tid - off], y = s1[tid];
        float mm = fmaxf(x, y);
        v = mm + log1pf(__expf(-fabsf(x - y)));
      }
      __syncthreads();
      if (act) s1[tid] = v;
      __syncthreads();
    }
    if (tid < U1) alpha[tid] = cval + s1[tid];
    __syncthreads();
    if (t == tcap && tid == 0) loss[b] = -(alpha[ucap] + blk[(size_t)t * U1 + ucap]);
    __syncthreads();
  }
}

__global__ void mean_kernel(const float* __restrict__ loss, float* __restrict__ out) {
  if (threadIdx.x == 0) {
    float s = 0.f;
    for (int i = 0; i < BB; ++i) s += loss[i];
    out[0] = s / (float)BB;
  }
}

// ---------------------------------------------------------------------------
extern "C" void kernel_launch(void* const* d_in, const int* in_sizes, int n_in,
                              void* d_out, int out_size, void* d_ws, size_t ws_size,
                              hipStream_t stream) {
  (void)in_sizes; (void)n_in; (void)out_size; (void)ws_size;
  const int*   enc_seqs = (const int*)d_in[0];
  const int*   enc_lens = (const int*)d_in[2];
  const int*   dec_seqs = (const int*)d_in[5];
  const int*   dec_lens = (const int*)d_in[7];
  const float* enc_Wk   = (const float*)d_in[10];
  const float* enc_Wr   = (const float*)d_in[11];
  const float* enc_b    = (const float*)d_in[12];
  const float* dec_Wk   = (const float*)d_in[13];
  const float* dec_Wr   = (const float*)d_in[14];
  const float* dec_b    = (const float*)d_in[15];
  const float* joint_W  = (const float*)d_in[16];
  const float* joint_b  = (const float*)d_in[17];
  float* out = (float*)d_out;

  char* p = (char*)d_ws;
  auto alloc = [&](size_t bytes) -> void* {
    void* r = (void*)p;
    p += (bytes + 255) & ~(size_t)255;
    return r;
  };
  unsigned*  ctrl0  = (unsigned*)alloc(64);
  unsigned*  ctrl1  = (unsigned*)alloc(64);
  float*     lossb  = (float*)alloc(BB * sizeof(float));
  _Float16*  WrtE   = (_Float16*)alloc((size_t)G4 * HH * 2);
  _Float16*  WrtD   = (_Float16*)alloc((size_t)G4 * HH * 2);
  _Float16*  Wet    = (_Float16*)alloc((size_t)VJ * HH * 2);
  _Float16*  Wdt    = (_Float16*)alloc((size_t)VJ * HH * 2);
  float*     zxE    = (float*)alloc((size_t)BB * TT * G4 * 4);
  float*     zxD    = (float*)alloc((size_t)BB * U1 * G4 * 4);
  float*     hstate = (float*)alloc((size_t)BB * HH * 4);
  float*     cstate = (float*)alloc((size_t)BB * HH * 4);
  _Float16*  hbuf   = (_Float16*)alloc((size_t)2 * BB * HH * 2);
  _Float16*  hsE    = (_Float16*)alloc((size_t)BB * TT * HH * 2);
  _Float16*  hsD    = (_Float16*)alloc((size_t)BB * U1 * HH * 2);
  float*     Ebuf   = (float*)alloc((size_t)BB * TT * VJ * 4);
  float*     Dbuf   = (float*)alloc((size_t)BB * U1 * VJ * 4);
  float*     blankB = (float*)alloc((size_t)BB * TT * U1 * 4);
  float*     emitB  = (float*)alloc((size_t)BB * TT * UU * 4);

  hipMemsetAsync(ctrl0, 0, 64, stream);
  hipMemsetAsync(ctrl1, 0, 64, stream);

  f16t_kernel<<<(G4 * HH + 255) / 256, 256, 0, stream>>>(enc_Wr, WrtE, HH, G4);
  f16t_kernel<<<(G4 * HH + 255) / 256, 256, 0, stream>>>(dec_Wr, WrtD, HH, G4);
  f16t_kernel<<<(VJ * HH + 255) / 256, 256, 0, stream>>>(joint_W, Wet, HH, VJ);
  f16t_kernel<<<(VJ * HH + 255) / 256, 256, 0, stream>>>(joint_W + (size_t)HH * VJ, Wdt, HH, VJ);
  zx_kernel<<<BB * TT, 256, 0, stream>>>(enc_seqs, enc_Wk, enc_b, zxE, TT, 0);
  zx_kernel<<<BB * U1, 256, 0, stream>>>(dec_seqs, dec_Wk, dec_b, zxD, U1, 1);

  lstm_kernel<<<BB, 256, 0, stream>>>(zxE, WrtE, enc_lens, 0, TT,
                                      hstate, cstate, hbuf, hsE, ctrl0);
  lstm_kernel<<<BB, 256, 0, stream>>>(zxD, WrtD, dec_lens, 1, U1,
                                      hstate, cstate, hbuf, hsD, ctrl1);

  proj_kernel<<<(BB * TT) / 16, 256, 0, stream>>>(hsE, Wet, nullptr, Ebuf, BB * TT);
  proj_kernel<<<(BB * U1) / 16, 256, 0, stream>>>(hsD, Wdt, joint_b, Dbuf, BB * U1);

  joint_kernel<<<BB * TT, 128, 0, stream>>>(Ebuf, Dbuf, dec_seqs, blankB, emitB);
  alpha_kernel<<<BB, 128, 0, stream>>>(blankB, emitB, enc_lens, dec_lens, lossb);
  mean_kernel<<<1, 32, 0, stream>>>(lossb, out);
}
